// MRNN_91199335563544
// MI455X (gfx1250) — matrix operand order repaired and checked by reading, not compiled, not run on hardware
//
#include <hip/hip_runtime.h>
#include <math.h>

#define DEV __device__ __forceinline__

typedef __attribute__((ext_vector_type(16))) __bf16        v16bf;
typedef __attribute__((ext_vector_type(8)))  float         v8f;
typedef __attribute__((ext_vector_type(4)))  unsigned int  v4u;
typedef __attribute__((ext_vector_type(8)))  int           v8i;
typedef __attribute__((ext_vector_type(4)))  int           v4i;
typedef __attribute__((ext_vector_type(4)))  unsigned short us4;

constexpr int T_STEPS = 2048;
constexpr int BATCH   = 64;
constexpr int FEAT    = 512;
constexpr int HID     = 512;
constexpr int NWG     = 16;            // persistent workgroups
constexpr int HS      = HID / NWG;     // 32 hidden columns per WG
constexpr int THREADS = 256;           // 8 wave32s
constexpr int LSTR    = FEAT + 8;      // padded LDS row stride (ushort); 1040B = 65*16, keeps 16B alignment

// LDS partition (ushort element offsets)
constexpr int OFF_X   = 0;                  // x_t    : 64 x LSTR (bf16)
constexpr int OFF_H   = BATCH * LSTR;       // h_{t-1}: 64 x LSTR (bf16)
constexpr int OFF_WIH = 2 * BATCH * LSTR;   // W_ih slice (i_n rows 0..31, i_f rows 32..63)
constexpr int OFF_WHH = 3 * BATCH * LSTR;   // W_hh slice: 32 x LSTR
constexpr int SMEM_US = 3 * BATCH * LSTR + HS * LSTR;
constexpr size_t SMEM_BYTES = (size_t)SMEM_US * sizeof(unsigned short);

// workspace layout (bytes)
constexpr size_t WS_CTR   = 0;              // barrier counter (zeroed each launch)
constexpr size_t WS_HBUF  = 256;            // two bf16 h buffers, 64x512x2 = 64KB each
constexpr size_t HB_BYTES = (size_t)BATCH * HID * 2;

DEV unsigned short f2b(float f) {            // native v_cvt f32->bf16 (RNE)
  union { __bf16 b; unsigned short u; } x; x.b = (__bf16)f; return x.u;
}
DEV float b2f(unsigned short s) {
  union { unsigned short u; __bf16 b; } x; x.u = s; return (float)x.b;
}

union V16 { v16bf v; unsigned u[8]; uint4 q[2]; };

// A-matrix 16x32 bf16 wave32 layout: two contiguous 16B runs per lane -> 2x ds_load_b128
DEV v16bf load_a(const unsigned short* base, int tile_m, int kk, int lane) {
  V16 r;
  const int m  = tile_m * 16 + (lane & 15);
  const int kb = kk * 32 + ((lane >> 4) << 3);
  const unsigned short* p = base + m * LSTR + kb;
  r.q[0] = *(const uint4*)(p);        // K pairs kb+0..kb+7
  r.q[1] = *(const uint4*)(p + 16);   // K pairs kb+16..kb+23
  return r.v;
}

// B-matrix 32x16 bf16 wave32 layout: two contiguous 16B runs per lane (same K order as A) -> 2x ds_load_b128
DEV v16bf load_b(const unsigned short* base, int rowbase, int kk, int lane) {
  V16 r;
  const int n  = rowbase + (lane & 15);
  const int kb = kk * 32 + ((lane >> 4) << 3);
  const unsigned short* p = base + n * LSTR + kb;
  r.q[0] = *(const uint4*)(p);
  r.q[1] = *(const uint4*)(p + 16);
  return r.v;
}

DEV void cvt_store4(unsigned short* d, float4 v) {   // 4x cvt + one ds_store_b64
  us4 s; s.x = f2b(v.x); s.y = f2b(v.y); s.z = f2b(v.z); s.w = f2b(v.w);
  *(us4*)d = s;
}

__global__ __launch_bounds__(THREADS)
void mrnn_scan(const float* __restrict__ x, const float* __restrict__ wih,
               const float* __restrict__ whh, const float* __restrict__ bih,
               const float* __restrict__ bhh, float* __restrict__ out,
               unsigned char* __restrict__ ws)
{
  extern __shared__ unsigned short smem[];
  unsigned short* lds_x   = smem + OFF_X;
  unsigned short* lds_h   = smem + OFF_H;
  unsigned short* lds_wih = smem + OFF_WIH;
  unsigned short* lds_whh = smem + OFF_WHH;

  unsigned int*   ctr   = (unsigned int*)(ws + WS_CTR);
  unsigned short* hbuf0 = (unsigned short*)(ws + WS_HBUF);
  unsigned short* hbuf1 = (unsigned short*)(ws + WS_HBUF + HB_BYTES);

  const int wg   = blockIdx.x;
  const int tid  = threadIdx.x;
  const int lane = tid & 31;
  const int wave = tid >> 5;
  const int tile_m = wave >> 1;   // 0..3 : batch rows 16*tile_m
  const int tile_n = wave & 1;    // 0..1 : hidden cols 16*tile_n within slice

  // ---- one-time: stage bf16 weight slices into LDS (float4 -> ushort4) ----
  for (int q = tid; q < 2 * HS * FEAT / 4; q += THREADS) {
    int e = q * 4, row = e >> 9, col = e & 511;
    int g = (row < HS) ? (wg * HS + row) : (HID + wg * HS + (row - HS));
    float4 v = *(const float4*)(wih + (size_t)g * FEAT + col);
    cvt_store4(lds_wih + row * LSTR + col, v);
  }
  for (int q = tid; q < HS * HID / 4; q += THREADS) {
    int e = q * 4, row = e >> 9, col = e & 511;
    float4 v = *(const float4*)(whh + (size_t)(wg * HS + row) * HID + col);
    cvt_store4(lds_whh + row * LSTR + col, v);
  }

  // per-lane constants for the elementwise gate phase (C/D layout)
  const int ncol  = tile_n * 16 + (lane & 15);
  const int gcol  = wg * HS + ncol;              // global hidden column
  const float bin = bih[gcol];
  const float bif = bih[HID + gcol];
  const float bhg = bhh[gcol];
  const int mrow0 = tile_m * 16 + ((lane >> 4) << 3);

  // ---- prologue: stage x_0 and compute its input-GEMM tiles ----
  __syncthreads();
  for (int q = tid; q < BATCH * FEAT / 4; q += THREADS) {
    int e = q * 4, row = e >> 9, col = e & 511;
    float4 v = *(const float4*)(x + (row << 9) + col);
    cvt_store4(lds_x + row * LSTR + col, v);
  }
  __syncthreads();

  v8f acc_n = {}, acc_f = {};
#pragma unroll 4
  for (int kk = 0; kk < FEAT / 32; ++kk) {
    v16bf a_x = load_a(lds_x, tile_m, kk, lane);
    v16bf b_n = load_b(lds_wih,             tile_n * 16, kk, lane);
    v16bf b_f = load_b(lds_wih + HS * LSTR, tile_n * 16, kk, lane);
    acc_n = __builtin_amdgcn_wmma_f32_16x16x32_bf16(false, a_x, false, b_n, (short)0, acc_n, false, false);
    acc_f = __builtin_amdgcn_wmma_f32_16x16x32_bf16(false, a_x, false, b_f, (short)0, acc_f, false, false);
  }

  for (int t = 0; t < T_STEPS; ++t) {
    // ---- stage h_{t-1} -> bf16 LDS: TDM (rows 0..31) + async-b128 (rows 32..63) ----
    if (t == 0) {
      for (int q = tid; q < BATCH * LSTR; q += THREADS) lds_h[q] = 0;
    } else {
      const unsigned short* src = ((t - 1) & 1) ? hbuf1 : hbuf0;
      unsigned long long sbase = (unsigned long long)(size_t)src;
      unsigned lbase = (unsigned)(unsigned long long)(size_t)lds_h;  // LDS byte addr (low 32 bits of shared-aperture ptr)

      if (wave == 0) {
        // Tensor DMA descriptor: 2D tile 512x32 of 2B elems, LDS pad 4 DWORDs per 256 DWORDs (row 1024B -> stride 1040B)
        v4u g0;
        g0[0] = 1u;                                               // count=1, user mode
        g0[1] = lbase;                                            // lds_addr
        g0[2] = (unsigned)(sbase & 0xFFFFFFFFu);                  // global_addr[31:0]
        g0[3] = (unsigned)((sbase >> 32) & 0x1FFFFFFu) | (2u << 30); // global_addr[56:32] | type=2
        v8i g1;
        g1[0] = (int)((1u << 16) | (1u << 20) | (7u << 22) | (3u << 25)); // data_size=2B, pad_en, interval=256dw, amount=4dw
        g1[1] = (int)(((unsigned)FEAT & 0xFFFFu) << 16);          // tensor_dim0[15:0]=512
        g1[2] = (int)(((unsigned)FEAT >> 16) | ((32u & 0xFFFFu) << 16)); // tensor_dim1[15:0]=32
        g1[3] = (int)((32u >> 16) | ((unsigned)FEAT << 16));      // tile_dim0=512
        g1[4] = (int)(32u);                                       // tile_dim1=32, tile_dim2=0
        g1[5] = (int)FEAT;                                        // tensor_dim0_stride=512 elems
        g1[6] = 0;
        g1[7] = 0;
        v4i gz = {0, 0, 0, 0};
        v8i gz8 = {0, 0, 0, 0, 0, 0, 0, 0};
        __builtin_amdgcn_tensor_load_to_lds(g0, g1, gz, gz, gz8, 0);
      }
      // rows 32..63 via per-lane async copies (ASYNCcnt), 16B each
      for (int q = tid; q < (BATCH / 2) * HID / 8; q += THREADS) {
        int e = q * 8, row = 32 + (e >> 9), col = e & 511;
        unsigned goff = (unsigned)((row << 10) + (col << 1));
        unsigned loff = lbase + (unsigned)(row * (LSTR * 2) + (col << 1));
        asm volatile("global_load_async_to_lds_b128 %0, %1, %2 offset:0"
                     :: "v"(loff), "v"(goff), "s"(sbase) : "memory");
      }
      asm volatile("s_wait_asynccnt 0x0" ::: "memory");
      if (wave == 0) __builtin_amdgcn_s_wait_tensorcnt(0);
    }
    __syncthreads();

    // ---- recurrent GEMM (critical path): gh tile ----
    v8f acc_g = {};
#pragma unroll 4
    for (int kk = 0; kk < HID / 32; ++kk) {
      v16bf a_h = load_a(lds_h, tile_m, kk, lane);
      v16bf b_g = load_b(lds_whh, tile_n * 16, kk, lane);
      acc_g = __builtin_amdgcn_wmma_f32_16x16x32_bf16(false, a_h, false, b_g, (short)0, acc_g, false, false);
    }

    // ---- gates + state update; fp32 h_t -> out, bf16 h_t slice -> workspace ----
    float* ot = out + (size_t)t * BATCH * HID;
    unsigned short* hw = (t & 1) ? hbuf1 : hbuf0;
#pragma unroll
    for (int r = 0; r < 8; ++r) {
      int brow = mrow0 + r;
      float hprev = b2f(lds_h[brow * LSTR + gcol]);        // zeros at t==0
      float z = acc_f[r] + bif + acc_g[r] + bhg;
      float fg = 1.0f / (1.0f + __expf(-z));
      float hn = (1.0f - fg) * (acc_n[r] + bin) + fg * hprev;
      ot[brow * HID + gcol] = hn;                          // lanes 0-15 contiguous
      hw[(brow << 9) + gcol] = f2b(hn);
    }

    // ---- signal grid barrier ----
    __threadfence();
    __syncthreads();
    const bool clustered = (__builtin_amdgcn_cluster_id_x() != 0);
    if (!clustered && tid == 0)
      __hip_atomic_fetch_add(ctr, 1u, __ATOMIC_ACQ_REL, __HIP_MEMORY_SCOPE_AGENT);

    // ---- overlap: stage x_{t+1} and its input-GEMM tiles behind the barrier ----
    if (t + 1 < T_STEPS) {
      const float* xt = x + (size_t)(t + 1) * BATCH * FEAT;
      for (int q = tid; q < BATCH * FEAT / 4; q += THREADS) {
        int e = q * 4, row = e >> 9, col = e & 511;
        float4 v = *(const float4*)(xt + (row << 9) + col);
        cvt_store4(lds_x + row * LSTR + col, v);
      }
      if (t + 2 < T_STEPS)   // warm L2 for the step after (global_prefetch_b8)
        __builtin_prefetch(x + (size_t)(t + 2) * BATCH * FEAT + (size_t)tid * 128, 0, 1);
      __syncthreads();

      acc_n = (v8f){}; acc_f = (v8f){};
#pragma unroll 4
      for (int kk = 0; kk < FEAT / 32; ++kk) {
        v16bf a_x = load_a(lds_x, tile_m, kk, lane);
        v16bf b_n = load_b(lds_wih,             tile_n * 16, kk, lane);
        v16bf b_f = load_b(lds_wih + HS * LSTR, tile_n * 16, kk, lane);
        acc_n = __builtin_amdgcn_wmma_f32_16x16x32_bf16(false, a_x, false, b_n, (short)0, acc_n, false, false);
        acc_f = __builtin_amdgcn_wmma_f32_16x16x32_bf16(false, a_x, false, b_f, (short)0, acc_f, false, false);
      }
    }

    // ---- wait on grid barrier ----
    if (clustered) {
      __builtin_amdgcn_s_cluster_barrier();                // fast path if dispatched as a cluster
    } else {
      if (tid == 0) {
        unsigned target = (unsigned)NWG * (unsigned)(t + 1);
        while (__hip_atomic_load(ctr, __ATOMIC_ACQUIRE, __HIP_MEMORY_SCOPE_AGENT) < target)
          __builtin_amdgcn_s_sleep(1);
      }
      __syncthreads();
    }
  }
}

extern "C" void kernel_launch(void* const* d_in, const int* in_sizes, int n_in,
                              void* d_out, int out_size, void* d_ws, size_t ws_size,
                              hipStream_t stream) {
  const float* x   = (const float*)d_in[0];   // [T,B,F]
  const float* wih = (const float*)d_in[1];   // [2H,F]
  const float* whh = (const float*)d_in[2];   // [H,H]
  const float* bih = (const float*)d_in[3];   // [2H]
  const float* bhh = (const float*)d_in[4];   // [H]
  float* out = (float*)d_out;                 // [T,B,H]
  unsigned char* ws = (unsigned char*)d_ws;

  (void)hipFuncSetAttribute(reinterpret_cast<const void*>(mrnn_scan),
                            hipFuncAttributeMaxDynamicSharedMemorySize, (int)SMEM_BYTES);
  (void)hipMemsetAsync(d_ws, 0, 256, stream); // reset barrier counter (capture-safe)
  mrnn_scan<<<NWG, THREADS, SMEM_BYTES, stream>>>(x, wih, whh, bih, bhh, out, ws);
}
